// GINModel_81827716923657
// MI455X (gfx1250) — matrix operand order repaired and checked by reading, not compiled, not run on hardware
//
#include <hip/hip_runtime.h>

// ---------------------------------------------------------------------------
// GIN model for MI455X (gfx1250): bf16 activations + v_wmma_f32_16x16x32_bf16
// GEMMs (4 N-tiles register-blocked per wave), f32 atomic scatter-sum for the
// graph aggregation.
// ---------------------------------------------------------------------------

typedef __attribute__((ext_vector_type(16))) __bf16 v16bf;
typedef __attribute__((ext_vector_type(8)))  float  v8f;

#define NB      100000     // nodes
#define NE      3200000    // edges
#define NG      2048       // graphs
#define F_IN    78
#define F_PAD   96         // 78 padded to multiple of 32 for WMMA K
#define OUTD    128

__device__ __forceinline__ unsigned short f2bf(float f) {
  unsigned int u = __float_as_uint(f);
  u += 0x7FFFu + ((u >> 16) & 1u);               // round-to-nearest-even
  return (unsigned short)(u >> 16);
}
__device__ __forceinline__ float bf2f(unsigned short h) {
  return __uint_as_float(((unsigned int)h) << 16);
}
__device__ __forceinline__ float elu1(float v) {
  return v > 0.f ? v : (__expf(v) - 1.f);
}

// ---------------------------------------------------------------------------
// Elementwise helpers
// ---------------------------------------------------------------------------
__global__ void zero_f32_kernel(float4* __restrict__ p, long long n4) {
  long long i = (long long)blockIdx.x * blockDim.x + threadIdx.x;
  if (i < n4) p[i] = make_float4(0.f, 0.f, 0.f, 0.f);
}

// x (f32, [NB,F_IN]) -> bf16 [NB,F_PAD] zero-padded
__global__ void cvt_x_pad_kernel(const float* __restrict__ x,
                                 unsigned short* __restrict__ out) {
  long long i = (long long)blockIdx.x * blockDim.x + threadIdx.x;
  long long n = (long long)NB * F_PAD;
  if (i >= n) return;
  int r = (int)(i / F_PAD), c = (int)(i % F_PAD);
  out[i] = (c < F_IN) ? f2bf(x[(size_t)r * F_IN + c]) : (unsigned short)0;
}

// sum_bf = bf16( bf2f(h_in) + agg )
__global__ void sum_bf_kernel(const unsigned short* __restrict__ hin,
                              const float* __restrict__ agg,
                              unsigned short* __restrict__ out, long long n) {
  long long i = (long long)blockIdx.x * blockDim.x + threadIdx.x;
  if (i < n) out[i] = f2bf(bf2f(hin[i]) + agg[i]);
}

// ---------------------------------------------------------------------------
// Weight pack: f32 [Korig,N] -> bf16 pairs in WMMA B-fragment order.
// Layout: uint index = ((kt*ntiles + nt)*32 + lane)*8 + v
//   n       = nt*16 + (lane&15)
//   k_local = (v<4 ? 2*v : 16 + 2*(v-4)) + (lane>>4)*8  (pair: k_local, k_local+1)   [ISA 7.12.2]
// K is padded to a multiple of 32 with zeros.
// ---------------------------------------------------------------------------
__global__ void pack_w_kernel(const float* __restrict__ W,
                              unsigned int* __restrict__ Wp,
                              int Korig, int K, int N) {
  int idx = blockIdx.x * blockDim.x + threadIdx.x;
  int total = (K >> 1) * N;
  if (idx >= total) return;
  int v    = idx & 7;
  int lane = (idx >> 3) & 31;
  int rest = idx >> 8;                 // / (8*32)
  int ntiles = N >> 4;
  int nt = rest % ntiles;
  int kt = rest / ntiles;
  int n  = nt * 16 + (lane & 15);
  int kl = ((v < 4) ? (2 * v) : (16 + 2 * (v - 4))) + ((lane >> 4) << 3);
  int k0 = kt * 32 + kl, k1 = k0 + 1;
  unsigned short lo = (k0 < Korig) ? f2bf(W[(size_t)k0 * N + n]) : (unsigned short)0;
  unsigned short hi = (k1 < Korig) ? f2bf(W[(size_t)k1 * N + n]) : (unsigned short)0;
  Wp[idx] = ((unsigned int)hi << 16) | lo;
}

// ---------------------------------------------------------------------------
// Edge scatter: agg[dst,f] += bf2f(h[src,f]); one thread per (edge, 4 feats).
// Consecutive threads cover one source row contiguously (coalesced 8B loads).
// Templated on D so t/(D/4) is a constant division.
// ---------------------------------------------------------------------------
template <int D>
__global__ void edge_scatter_kernel(const int* __restrict__ ei,
                                    const unsigned short* __restrict__ h,
                                    float* __restrict__ agg) {
  constexpr int PER = D >> 2;
  long long t = (long long)blockIdx.x * blockDim.x + threadIdx.x;
  long long total = (long long)NE * PER;
  if (t >= total) return;
  int e  = (int)(t / PER);
  int fo = ((int)(t % PER)) << 2;
  int src = ei[e];
  int dst = ei[NE + e];
  const uint2 pv = *reinterpret_cast<const uint2*>(h + (size_t)src * D + fo);
  float* q = agg + (size_t)dst * D + fo;
  atomicAdd(q + 0, bf2f((unsigned short)(pv.x & 0xFFFFu)));
  atomicAdd(q + 1, bf2f((unsigned short)(pv.x >> 16)));
  atomicAdd(q + 2, bf2f((unsigned short)(pv.y & 0xFFFFu)));
  atomicAdd(q + 3, bf2f((unsigned short)(pv.y >> 16)));
}

// ---------------------------------------------------------------------------
// WMMA GEMM: out = elu^nElu( A[M,K]bf16 @ W[K,N]bf16 + bias ).
// M % 16 == 0, K % 32 == 0, N % 64 == 0. Block = 128 thr = 4 waves; each wave
// owns a 16-row tile and register-blocks 4 N-tiles (4 independent accumulator
// chains -> hides load latency and WMMA->WMMA hazards, 4x fewer A reloads).
// ---------------------------------------------------------------------------
template <int K, int N>
__global__ void gemm_bf16_kernel(const unsigned short* __restrict__ A,
                                 const unsigned int* __restrict__ Wp,
                                 const float* __restrict__ bias,
                                 float* __restrict__ outF,
                                 unsigned short* __restrict__ outBF,
                                 int M, int nElu) {
  constexpr int KT = K >> 5;           // K-steps of 32
  constexpr int NTILES = N >> 4;       // 16-wide N tiles (multiple of 4)
  const int lane = threadIdx.x & 31;
  const int wave = threadIdx.x >> 5;
  const int mbase = blockIdx.x * 64 + wave * 16;
  if (mbase >= M) return;
  const int half = lane >> 4;          // 0: lanes 0-15, 1: lanes 16-31
  const int lm   = lane & 15;
  const unsigned int* __restrict__ Arow =
      reinterpret_cast<const unsigned int*>(A) + (size_t)(mbase + lm) * (K >> 1);

  for (int nt0 = 0; nt0 < NTILES; nt0 += 4) {
    v8f acc[4] = {{}, {}, {}, {}};
#pragma unroll
    for (int kt = 0; kt < KT; ++kt) {
      // A fragment (16-bit A 16x32 lane layout, ISA 7.12.2), shared by 4 tiles
      union { unsigned int u[8]; v16bf v; } a;
      const int kb = kt * 32;
#pragma unroll
      for (int v = 0; v < 8; ++v) {
        int kl = (v < 4) ? (2 * v + 8 * half) : (16 + 2 * (v - 4) + 8 * half);
        a.u[v] = Arow[(kb + kl) >> 1];
      }
#pragma unroll
      for (int i = 0; i < 4; ++i) {
        union { unsigned int u[8]; v16bf v; } b;
        const unsigned int* __restrict__ wl =
            Wp + (((size_t)kt * NTILES + nt0 + i) * 32 + lane) * 8;
#pragma unroll
        for (int v = 0; v < 8; ++v) b.u[v] = wl[v];
        acc[i] = __builtin_amdgcn_wmma_f32_16x16x32_bf16(
            false, a.v, false, b.v, (short)0, acc[i], false, false);
      }
    }
    // C/D fragment layout: VGPR j -> row mbase + j + 8*half, col nt*16 + lm
#pragma unroll
    for (int i = 0; i < 4; ++i) {
      const int col = (nt0 + i) * 16 + lm;
      const float bv = bias[col];
#pragma unroll
      for (int j = 0; j < 8; ++j) {
        int r = mbase + j + 8 * half;
        float vv = acc[i][j] + bv;
        for (int t = 0; t < nElu; ++t) vv = elu1(vv);
        if (outF)  outF[(size_t)r * N + col]  = vv;
        if (outBF) outBF[(size_t)r * N + col] = f2bf(vv);
      }
    }
  }
}

// ---------------------------------------------------------------------------
// Mean pool over graphs
// ---------------------------------------------------------------------------
__global__ void pool_count_kernel(const int* __restrict__ batch,
                                  float* __restrict__ counts) {
  int i = blockIdx.x * blockDim.x + threadIdx.x;
  if (i < NB) atomicAdd(&counts[batch[i]], 1.f);
}
__global__ void pool_sum_kernel(const int* __restrict__ batch,
                                const unsigned short* __restrict__ h,
                                float* __restrict__ pooled) {
  long long t = (long long)blockIdx.x * blockDim.x + threadIdx.x;
  long long total = (long long)NB * (OUTD >> 2);
  if (t >= total) return;
  int node = (int)(t / (OUTD >> 2));
  int fo   = ((int)(t % (OUTD >> 2))) << 2;
  int b = batch[node];
  const uint2 pv = *reinterpret_cast<const uint2*>(h + (size_t)node * OUTD + fo);
  float* q = pooled + (size_t)b * OUTD + fo;
  atomicAdd(q + 0, bf2f((unsigned short)(pv.x & 0xFFFFu)));
  atomicAdd(q + 1, bf2f((unsigned short)(pv.x >> 16)));
  atomicAdd(q + 2, bf2f((unsigned short)(pv.y & 0xFFFFu)));
  atomicAdd(q + 3, bf2f((unsigned short)(pv.y >> 16)));
}
__global__ void pool_div_kernel(const float* __restrict__ pooled,
                                const float* __restrict__ counts,
                                unsigned short* __restrict__ gbf) {
  int i = blockIdx.x * blockDim.x + threadIdx.x;
  int n = NG * OUTD;
  if (i >= n) return;
  float c = counts[i / OUTD];
  gbf[i] = f2bf(pooled[i] / fmaxf(c, 1.f));
}

// ---------------------------------------------------------------------------
// Host orchestration
// ---------------------------------------------------------------------------
extern "C" void kernel_launch(void* const* d_in, const int* in_sizes, int n_in,
                              void* d_out, int out_size, void* d_ws, size_t ws_size,
                              hipStream_t stream) {
  const float* x    = (const float*)d_in[0];
  const int*   ei   = (const int*)d_in[1];
  const int*   batch= (const int*)d_in[2];
  const float *w1a=(const float*)d_in[3],  *b1a=(const float*)d_in[4];
  const float *w1b=(const float*)d_in[5],  *b1b=(const float*)d_in[6];
  const float *w2a=(const float*)d_in[7],  *b2a=(const float*)d_in[8];
  const float *w2b=(const float*)d_in[9],  *b2b=(const float*)d_in[10];
  const float *w3a=(const float*)d_in[11], *b3a=(const float*)d_in[12];
  const float *w3b=(const float*)d_in[13], *b3b=(const float*)d_in[14];
  const float *wm1=(const float*)d_in[15], *bm1=(const float*)d_in[16];
  const float *wm2=(const float*)d_in[17], *bm2=(const float*)d_in[18];
  const float *wm3=(const float*)d_in[19], *bm3=(const float*)d_in[20];

  char* ws = (char*)d_ws;
  size_t off = 0;
  auto alloc = [&](size_t bytes) -> void* {
    off = (off + 255) & ~(size_t)255;
    void* p = ws + off;
    off += bytes;
    return p;
  };
  unsigned short* bfA = (unsigned short*)alloc((size_t)NB * 256 * 2);
  unsigned short* bfB = (unsigned short*)alloc((size_t)NB * 256 * 2);
  unsigned short* bfC = (unsigned short*)alloc((size_t)NB * 256 * 2);
  float*          agg = (float*)alloc((size_t)NB * 256 * 4);
  unsigned int* p1a = (unsigned int*)alloc((size_t)F_PAD * 128 * 2);
  unsigned int* p1b = (unsigned int*)alloc((size_t)128 * 128 * 2);
  unsigned int* p2a = (unsigned int*)alloc((size_t)128 * 256 * 2);
  unsigned int* p2b = (unsigned int*)alloc((size_t)256 * 256 * 2);
  unsigned int* p3a = (unsigned int*)alloc((size_t)256 * 128 * 2);
  unsigned int* p3b = (unsigned int*)alloc((size_t)128 * 128 * 2);
  unsigned int* pm1 = (unsigned int*)alloc((size_t)128 * 256 * 2);
  unsigned int* pm2 = (unsigned int*)alloc((size_t)256 * 128 * 2);
  unsigned int* pm3 = (unsigned int*)alloc((size_t)128 * 128 * 2);
  float* pooled = (float*)alloc((size_t)NG * OUTD * 4);
  float* counts = (float*)alloc((size_t)NG * 4);
  unsigned short* g0 = (unsigned short*)alloc((size_t)NG * 128 * 2);
  unsigned short* g1 = (unsigned short*)alloc((size_t)NG * 256 * 2);
  unsigned short* g2 = (unsigned short*)alloc((size_t)NG * 128 * 2);

  auto gridFor = [](long long n, int blk) { return (unsigned)((n + blk - 1) / blk); };
  auto pack = [&](const float* W, unsigned int* Wp, int Korig, int K, int N) {
    int total = (K >> 1) * N;
    pack_w_kernel<<<gridFor(total, 256), 256, 0, stream>>>(W, Wp, Korig, K, N);
  };
  auto zero = [&](float* p, long long n) {   // n multiple of 4
    zero_f32_kernel<<<gridFor(n >> 2, 256), 256, 0, stream>>>((float4*)p, n >> 2);
  };
  auto sumbf = [&](const unsigned short* hin, unsigned short* o, int D) {
    long long n = (long long)NB * D;
    sum_bf_kernel<<<gridFor(n, 256), 256, 0, stream>>>(hin, agg, o, n);
  };

  // Weight packing (re-done every call: deterministic, no caching)
  pack(w1a, p1a, F_IN, F_PAD, 128);
  pack(w1b, p1b, 128, 128, 128);
  pack(w2a, p2a, 128, 128, 256);
  pack(w2b, p2b, 256, 256, 256);
  pack(w3a, p3a, 256, 256, 128);
  pack(w3b, p3b, 128, 128, 128);
  pack(wm1, pm1, 128, 128, 256);
  pack(wm2, pm2, 256, 256, 128);
  pack(wm3, pm3, 128, 128, 128);

  // x -> bf16 padded [NB, 96]
  cvt_x_pad_kernel<<<gridFor((long long)NB * F_PAD, 256), 256, 0, stream>>>(x, bfA);

  const unsigned gemmGrid = gridFor(NB, 64);
  const unsigned headGrid = gridFor(NG, 64);

  // ---- GIN layer 1 (Din=96 padded, 128 -> 128) ----
  zero(agg, (long long)NB * F_PAD);
  edge_scatter_kernel<F_PAD><<<gridFor((long long)NE * (F_PAD / 4), 256), 256, 0, stream>>>(ei, bfA, agg);
  sumbf(bfA, bfB, F_PAD);
  gemm_bf16_kernel<F_PAD, 128><<<gemmGrid, 128, 0, stream>>>(bfB, p1a, b1a, nullptr, bfC, NB, 1);
  gemm_bf16_kernel<128, 128><<<gemmGrid, 128, 0, stream>>>(bfC, p1b, b1b, nullptr, bfA, NB, 2);

  // ---- GIN layer 2 (128 -> 256 -> 256) ----
  zero(agg, (long long)NB * 128);
  edge_scatter_kernel<128><<<gridFor((long long)NE * 32, 256), 256, 0, stream>>>(ei, bfA, agg);
  sumbf(bfA, bfB, 128);
  gemm_bf16_kernel<128, 256><<<gemmGrid, 128, 0, stream>>>(bfB, p2a, b2a, nullptr, bfC, NB, 1);
  gemm_bf16_kernel<256, 256><<<gemmGrid, 128, 0, stream>>>(bfC, p2b, b2b, nullptr, bfA, NB, 2);

  // ---- GIN layer 3 (256 -> 128 -> 128, no trailing elu) ----
  zero(agg, (long long)NB * 256);
  edge_scatter_kernel<256><<<gridFor((long long)NE * 64, 256), 256, 0, stream>>>(ei, bfA, agg);
  sumbf(bfA, bfB, 256);
  gemm_bf16_kernel<256, 128><<<gemmGrid, 128, 0, stream>>>(bfB, p3a, b3a, nullptr, bfC, NB, 1);
  gemm_bf16_kernel<128, 128><<<gemmGrid, 128, 0, stream>>>(bfC, p3b, b3b, nullptr, bfA, NB, 1);

  // ---- global mean pool ----
  zero(pooled, (long long)NG * OUTD);
  zero(counts, NG);
  pool_count_kernel<<<gridFor(NB, 256), 256, 0, stream>>>(batch, counts);
  pool_sum_kernel<<<gridFor((long long)NB * (OUTD >> 2), 256), 256, 0, stream>>>(batch, bfA, pooled);
  pool_div_kernel<<<gridFor((long long)NG * OUTD, 256), 256, 0, stream>>>(pooled, counts, g0);

  // ---- head MLP ----
  gemm_bf16_kernel<128, 256><<<headGrid, 128, 0, stream>>>(g0, pm1, bm1, nullptr, g1, NG, 1);
  gemm_bf16_kernel<256, 128><<<headGrid, 128, 0, stream>>>(g1, pm2, bm2, nullptr, g2, NG, 1);
  gemm_bf16_kernel<128, 128><<<headGrid, 128, 0, stream>>>(g2, pm3, bm3, (float*)d_out, nullptr, NG, 1);
}
